// TokenPruner_10943576670975
// MI455X (gfx1250) — compile-verified, matrix operand order checked by reading, not run on hardware
//
#include <hip/hip_runtime.h>
#include <math.h>

// ---------------------------------------------------------------------------
// Problem constants (match the reference)
// ---------------------------------------------------------------------------
constexpr int L     = 4096;
constexpr int D     = 2048;
constexpr int NQ    = 16;
constexpr int NKV   = 8;
constexpr int HD    = 128;
constexpr int KSEL  = 1024;            // int(L * 0.25)
constexpr float SCALING = 0.08838834764831845f;   // 128^-0.5
constexpr float SOFTCAP = 50.0f;

typedef __attribute__((ext_vector_type(16))) _Float16 vh16;
typedef __attribute__((ext_vector_type(8)))  _Float16 vh8;
typedef __attribute__((ext_vector_type(8)))  float    vf8;
typedef __attribute__((ext_vector_type(4)))  int      vi4;

// ---------------------------------------------------------------------------
// CDNA5 async global->LDS copy (GLOBAL_LOAD_ASYNC_TO_LDS_B128, ASYNCcnt).
// Parameter types per hipcc diagnostic: (int4 AS1*, int4 AS3*, imm, cpol).
// Falls back to a synchronous LDS store if the builtin is unavailable.
// ---------------------------------------------------------------------------
#if __has_builtin(__builtin_amdgcn_global_load_async_to_lds_b128)
#define HAS_ASYNC_LDS 1
#else
#define HAS_ASYNC_LDS 0
#endif

__device__ __forceinline__ void copy16_async(const void* g, void* l)
{
#if HAS_ASYNC_LDS
    __builtin_amdgcn_global_load_async_to_lds_b128(
        (__attribute__((address_space(1))) vi4*)(unsigned long long)g,
        (__attribute__((address_space(3))) vi4*)l, 0, 0);
#else
    *(vh8*)l = *(const vh8*)g;
#endif
}

__device__ __forceinline__ void wait_async()
{
#if HAS_ASYNC_LDS
#if __has_builtin(__builtin_amdgcn_s_wait_asynccnt)
    __builtin_amdgcn_s_wait_asynccnt(0);
#else
    asm volatile("s_wait_asynccnt 0x0" ::: "memory");
#endif
#endif
}

// ---------------------------------------------------------------------------
// Weight transpose + f32->f16: W [K,N] f32 -> Wt [N,K] f16 (64x64 LDS tiles,
// coalesced both directions). Makes GEMM B-tiles contiguous-row, async-copy
// friendly (no per-element LDS transpose in the GEMM hot loop).
// ---------------------------------------------------------------------------
__global__ __launch_bounds__(256) void transpose_f16_kernel(
    const float* __restrict__ W, _Float16* __restrict__ Wt, int K, int N)
{
    __shared__ _Float16 tile[64][65];
    const int kb = blockIdx.x * 64;
    const int nb = blockIdx.y * 64;
    const int t  = threadIdx.x;
    #pragma unroll
    for (int i = 0; i < 16; ++i) {
        int e = t + i * 256;               // 0..4095
        int rk = e >> 6, cn = e & 63;
        tile[cn][rk] = (_Float16)W[(size_t)(kb + rk) * N + nb + cn];
    }
    __syncthreads();
    #pragma unroll
    for (int i = 0; i < 16; ++i) {
        int e = t + i * 256;
        int rn = e >> 6, ck = e & 63;
        Wt[(size_t)(nb + rn) * K + kb + ck] = tile[rn][ck];
    }
}

// ---------------------------------------------------------------------------
// Embedding gather -> f16  (out[l, d] = table[ids[l], d])
// ---------------------------------------------------------------------------
__global__ __launch_bounds__(256) void embed_gather_kernel(
    const int* __restrict__ ids, const float* __restrict__ table,
    _Float16* __restrict__ out)
{
    int l = blockIdx.x;
    int d = blockIdx.y * 256 + threadIdx.x;
    int tok = ids[l];
    out[(size_t)l * D + d] = (_Float16)table[(size_t)tok * D + d];
}

// ---------------------------------------------------------------------------
// f16 WMMA GEMM: C[M,N] (f32) = A[M,K] * Bt[N,K]^T, A/Bt row-major.
// 128x128 block tile, 8 waves (2x4), K-loop step 32, double-buffered LDS with
// async global->LDS staging.
// ---------------------------------------------------------------------------
__global__ __launch_bounds__(256) void gemm_f16_kernel(
    const _Float16* __restrict__ A, const _Float16* __restrict__ Bt,
    float* __restrict__ C, int N, int K)
{
    __shared__ __attribute__((aligned(16))) _Float16 sA[2][128][32];  // [m][k]
    __shared__ __attribute__((aligned(16))) _Float16 sB[2][128][32];  // [n][k]

    const int tid  = threadIdx.x;
    const int w    = tid >> 5;
    const int lane = tid & 31;
    const int half = lane >> 4;
    const int r    = lane & 15;
    const int m0   = blockIdx.x * 128;
    const int n0   = blockIdx.y * 128;
    const int wm   = (w >> 2) * 64;
    const int wn   = (w & 3) * 32;

    auto loadTile = [&](int b, int kc) {
        #pragma unroll
        for (int it = 0; it < 2; ++it) {
            int idx  = tid + it * 256;          // 0..511
            int row  = idx >> 2;
            int col8 = (idx & 3) * 8;
            copy16_async(&A [(size_t)(m0 + row) * K + kc + col8], &sA[b][row][col8]);
            copy16_async(&Bt[(size_t)(n0 + row) * K + kc + col8], &sB[b][row][col8]);
        }
    };

    vf8 acc[4][2] = {};

    loadTile(0, 0);
    wait_async();
    __syncthreads();

    int buf = 0;
    for (int kc = 0; kc < K; kc += 32) {
        if (kc + 32 < K) loadTile(buf ^ 1, kc + 32);   // prefetch next tile
        #pragma unroll
        for (int mi = 0; mi < 4; ++mi) {
            vh16 af;
            ((vh8*)&af)[0] = *(const vh8*)&sA[buf][wm + 16 * mi + r][half * 8];
            ((vh8*)&af)[1] = *(const vh8*)&sA[buf][wm + 16 * mi + r][16 + half * 8];
            #pragma unroll
            for (int ni = 0; ni < 2; ++ni) {
                vh16 bf;
                ((vh8*)&bf)[0] = *(const vh8*)&sB[buf][wn + 16 * ni + r][half * 8];
                ((vh8*)&bf)[1] = *(const vh8*)&sB[buf][wn + 16 * ni + r][16 + half * 8];
                acc[mi][ni] = __builtin_amdgcn_wmma_f32_16x16x32_f16(
                    false, af, false, bf, (short)0, acc[mi][ni], false, false);
            }
        }
        wait_async();
        __syncthreads();
        buf ^= 1;
    }

    #pragma unroll
    for (int mi = 0; mi < 4; ++mi)
        #pragma unroll
        for (int ni = 0; ni < 2; ++ni) {
            int mrow = m0 + wm + 16 * mi + 8 * half;
            int ncol = n0 + wn + 16 * ni + r;
            #pragma unroll
            for (int v = 0; v < 8; ++v)
                C[(size_t)(mrow + v) * N + ncol] = acc[mi][ni][v];
        }
}

// ---------------------------------------------------------------------------
// RoPE epilogue: f32 [L, nheads*HD] -> f16 with rotary applied per head.
// ---------------------------------------------------------------------------
__global__ __launch_bounds__(HD) void rope_kernel(
    const float* __restrict__ X, _Float16* __restrict__ Y, int nheads)
{
    int l = blockIdx.x;
    int h = blockIdx.y;
    int d = threadIdx.x;                         // 0..127
    size_t base = (size_t)l * nheads * HD + (size_t)h * HD;
    float x   = X[base + d];
    float rot = (d < 64) ? -X[base + d + 64] : X[base + d - 64];
    int   i2  = d & 63;
    float invf = expf(-(float)(2 * i2) / (float)HD * logf(10000.0f));
    float ang  = (float)l * invf;
    float y = x * cosf(ang) + rot * sinf(ang);
    Y[base + d] = (_Float16)y;
}

// ---------------------------------------------------------------------------
// Attention pass A: per (head, 128-query tile) online softmax row stats.
// Double-buffered async K tiles (sQ 32K + 2x16K sK = 64KB LDS).
// ---------------------------------------------------------------------------
__global__ __launch_bounds__(256) void attn_stats_kernel(
    const _Float16* __restrict__ Qh, const _Float16* __restrict__ Kh,
    float* __restrict__ rowM, float* __restrict__ rowZ)
{
    __shared__ __attribute__((aligned(16))) _Float16 sQ[128][HD];     // 32 KB
    __shared__ __attribute__((aligned(16))) _Float16 sK[2][64][HD];   // 32 KB

    const int q0   = blockIdx.x * 128;
    const int h    = blockIdx.y;
    const int kvh  = h >> 1;                     // GQA expand = 2
    const int tid  = threadIdx.x;
    const int w    = tid >> 5;
    const int lane = tid & 31;
    const int half = lane >> 4;
    const int r    = lane & 15;
    const int qb   = 16 * w;

    auto loadK = [&](int b, int k0) {
        #pragma unroll
        for (int it = 0; it < 4; ++it) {
            int c = tid + it * 256;              // 0..1023 chunks
            int row = c >> 4, col8 = (c & 15) * 8;
            copy16_async(&Kh[(size_t)(k0 + row) * (NKV * HD) + (size_t)kvh * HD + col8],
                         &sK[b][row][col8]);
        }
    };

    #pragma unroll
    for (int it = 0; it < 8; ++it) {
        int c = tid + it * 256;                  // 0..2047 chunks
        int row = c >> 4, col8 = (c & 15) * 8;
        copy16_async(&Qh[(size_t)(q0 + row) * (NQ * HD) + (size_t)h * HD + col8],
                     &sQ[row][col8]);
    }
    loadK(0, 0);
    wait_async();
    __syncthreads();

    float mrun[8], Zrun[8];
    #pragma unroll
    for (int v = 0; v < 8; ++v) { mrun[v] = -INFINITY; Zrun[v] = 0.0f; }

    const int nkt = q0 / 64 + 2;
    for (int kt = 0; kt < nkt; ++kt) {
        const int k0  = kt * 64;
        const int cur = kt & 1;
        if (kt + 1 < nkt) loadK(cur ^ 1, k0 + 64);   // async prefetch

        for (int ns = 0; ns < 4; ++ns) {
            vf8 acc = {};
            #pragma unroll
            for (int kc = 0; kc < 4; ++kc) {
                vh16 af, bf;
                ((vh8*)&af)[0] = *(const vh8*)&sQ[qb + r][kc * 32 + half * 8];
                ((vh8*)&af)[1] = *(const vh8*)&sQ[qb + r][kc * 32 + 16 + half * 8];
                ((vh8*)&bf)[0] = *(const vh8*)&sK[cur][ns * 16 + r][kc * 32 + half * 8];
                ((vh8*)&bf)[1] = *(const vh8*)&sK[cur][ns * 16 + r][kc * 32 + 16 + half * 8];
                acc = __builtin_amdgcn_wmma_f32_16x16x32_f16(
                    false, af, false, bf, (short)0, acc, false, false);
            }
            #pragma unroll
            for (int v = 0; v < 8; ++v) {
                int  qg = q0 + qb + v + 8 * half;
                int  kg = k0 + ns * 16 + r;
                bool valid = (kg <= qg);
                float sc = tanhf(acc[v] * (SCALING / SOFTCAP)) * SOFTCAP;
                float s  = valid ? sc : -INFINITY;
                float tm = s;
                tm = fmaxf(tm, __shfl_xor(tm, 1));
                tm = fmaxf(tm, __shfl_xor(tm, 2));
                tm = fmaxf(tm, __shfl_xor(tm, 4));
                tm = fmaxf(tm, __shfl_xor(tm, 8));
                float mnew  = fmaxf(mrun[v], tm);
                float msafe = (mnew == -INFINITY) ? 0.0f : mnew;
                float e = valid ? expf(sc - msafe) : 0.0f;
                e += __shfl_xor(e, 1);
                e += __shfl_xor(e, 2);
                e += __shfl_xor(e, 4);
                e += __shfl_xor(e, 8);
                float sOld = (mrun[v] == -INFINITY) ? 0.0f : expf(mrun[v] - msafe);
                Zrun[v] = Zrun[v] * sOld + e;
                mrun[v] = mnew;
            }
        }
        wait_async();
        __syncthreads();
    }

    if (r == 0) {
        #pragma unroll
        for (int v = 0; v < 8; ++v) {
            int qg = q0 + qb + v + 8 * half;
            rowM[(size_t)h * L + qg] = mrun[v];
            rowZ[(size_t)h * L + qg] = Zrun[v];
        }
    }
}

// ---------------------------------------------------------------------------
// Attention pass B: recompute scores, accumulate normalized column sums
// (importance) via LDS atomics + global atomic adds.
// ---------------------------------------------------------------------------
__global__ __launch_bounds__(256) void attn_accum_kernel(
    const _Float16* __restrict__ Qh, const _Float16* __restrict__ Kh,
    const float* __restrict__ rowM, const float* __restrict__ rowZ,
    float* __restrict__ imp)
{
    __shared__ __attribute__((aligned(16))) _Float16 sQ[128][HD];   // 32 KB
    __shared__ __attribute__((aligned(16))) _Float16 sK[64][HD];    // 16 KB
    __shared__ float sCol[64];

    const int q0   = blockIdx.x * 128;
    const int h    = blockIdx.y;
    const int kvh  = h >> 1;
    const int tid  = threadIdx.x;
    const int w    = tid >> 5;
    const int lane = tid & 31;
    const int half = lane >> 4;
    const int r    = lane & 15;
    const int qb   = 16 * w;
    const float invNorm = 1.0f / ((float)NQ * (float)L);

    float mrow[8], wZ[8];
    #pragma unroll
    for (int v = 0; v < 8; ++v) {
        int qg  = q0 + qb + v + 8 * half;
        mrow[v] = rowM[(size_t)h * L + qg];
        float Z = rowZ[(size_t)h * L + qg];
        wZ[v]   = (Z > 0.0f) ? invNorm / Z : 0.0f;
    }

    #pragma unroll
    for (int it = 0; it < 8; ++it) {
        int c = tid + it * 256;
        int row = c >> 4, col8 = (c & 15) * 8;
        copy16_async(&Qh[(size_t)(q0 + row) * (NQ * HD) + (size_t)h * HD + col8],
                     &sQ[row][col8]);
    }

    const int nkt = q0 / 64 + 2;
    for (int kt = 0; kt < nkt; ++kt) {
        const int k0 = kt * 64;
        if (tid < 64) sCol[tid] = 0.0f;
        #pragma unroll
        for (int it = 0; it < 4; ++it) {
            int c = tid + it * 256;
            int row = c >> 4, col8 = (c & 15) * 8;
            copy16_async(&Kh[(size_t)(k0 + row) * (NKV * HD) + (size_t)kvh * HD + col8],
                         &sK[row][col8]);
        }
        wait_async();
        __syncthreads();

        for (int ns = 0; ns < 4; ++ns) {
            vf8 acc = {};
            #pragma unroll
            for (int kc = 0; kc < 4; ++kc) {
                vh16 af, bf;
                ((vh8*)&af)[0] = *(const vh8*)&sQ[qb + r][kc * 32 + half * 8];
                ((vh8*)&af)[1] = *(const vh8*)&sQ[qb + r][kc * 32 + 16 + half * 8];
                ((vh8*)&bf)[0] = *(const vh8*)&sK[ns * 16 + r][kc * 32 + half * 8];
                ((vh8*)&bf)[1] = *(const vh8*)&sK[ns * 16 + r][kc * 32 + 16 + half * 8];
                acc = __builtin_amdgcn_wmma_f32_16x16x32_f16(
                    false, af, false, bf, (short)0, acc, false, false);
            }
            float cs = 0.0f;
            #pragma unroll
            for (int v = 0; v < 8; ++v) {
                int  qg = q0 + qb + v + 8 * half;
                int  kg = k0 + ns * 16 + r;
                bool valid = (kg <= qg);
                float sc = tanhf(acc[v] * (SCALING / SOFTCAP)) * SOFTCAP;
                float p  = valid ? expf(sc - mrow[v]) * wZ[v] : 0.0f;
                cs += p;
            }
            cs += __shfl_xor(cs, 16);            // merge both row halves
            if (lane < 16) atomicAdd(&sCol[ns * 16 + r], cs);  // ds_add_f32
        }
        __syncthreads();
        if (tid < 64) atomicAdd(&imp[k0 + tid], sCol[tid]);    // global_atomic_add_f32
        __syncthreads();
    }
}

// ---------------------------------------------------------------------------
// Importance zeroing (ws is poisoned once and never re-poisoned)
// ---------------------------------------------------------------------------
__global__ __launch_bounds__(256) void zero_f32_kernel(float* __restrict__ p, int n)
{
    int i = blockIdx.x * 256 + threadIdx.x;
    if (i < n) p[i] = 0.0f;
}

// ---------------------------------------------------------------------------
// Top-k selection by rank counting (jax top_k tie-break: value desc, idx asc)
// ---------------------------------------------------------------------------
__global__ __launch_bounds__(256) void rank_kernel(
    const float* __restrict__ imp, int* __restrict__ sel)
{
    __shared__ float simp[L];                    // 16 KB
    int t = threadIdx.x;
    for (int j = t; j < L; j += 256) simp[j] = imp[j];
    __syncthreads();
    int i = blockIdx.x * 256 + t;
    float vi = simp[i];
    int rank = 0;
    for (int j = 0; j < L; ++j) {
        float vj = simp[j];
        rank += (int)((vj > vi) | ((vj == vi) & (j < i)));
    }
    sel[i] = (rank < KSEL) ? 1 : 0;
}

// Single-block scan + emit sorted-ascending topk, append/pad L-1 to out_half.
// d_out layout: [pruned_ids (out_half)] [topk (out_half)], int32.
__global__ __launch_bounds__(256) void emit_kernel(
    const int* __restrict__ sel, const int* __restrict__ ids,
    int* __restrict__ out, int out_half)
{
    __shared__ int chunkSum[256];
    __shared__ int chunkOff[256];
    int t = threadIdx.x;
    int c0 = t * 16;
    int s = 0;
    for (int j = 0; j < 16; ++j) s += sel[c0 + j];
    chunkSum[t] = s;
    __syncthreads();
    if (t == 0) {
        int run = 0;
        for (int c = 0; c < 256; ++c) { chunkOff[c] = run; run += chunkSum[c]; }
    }
    __syncthreads();
    int pos = chunkOff[t];
    for (int j = 0; j < 16; ++j) {
        int i = c0 + j;
        if (sel[i]) {
            if (pos < out_half) { out[pos] = ids[i]; out[out_half + pos] = i; }
            ++pos;
        }
    }
    __syncthreads();
    if (t == 0) {
        int total = chunkOff[255] + chunkSum[255];   // == KSEL
        for (int p = total; p < out_half; ++p) {     // append L-1 (reference path)
            out[p]            = ids[L - 1];
            out[out_half + p] = L - 1;
        }
    }
}

// ---------------------------------------------------------------------------
// Launcher
// ---------------------------------------------------------------------------
extern "C" void kernel_launch(void* const* d_in, const int* in_sizes, int n_in,
                              void* d_out, int out_size, void* d_ws, size_t ws_size,
                              hipStream_t stream)
{
    (void)in_sizes; (void)n_in; (void)ws_size;
    const int*   ids   = (const int*)d_in[0];
    const float* table = (const float*)d_in[1];
    const float* wq    = (const float*)d_in[2];
    const float* wk    = (const float*)d_in[3];

    char* ws = (char*)d_ws;
    size_t off = 0;
    auto alloc = [&](size_t bytes) -> void* {
        void* p = ws + off;
        off += (bytes + 255) & ~(size_t)255;
        return p;
    };
    _Float16* Eh   = (_Float16*)alloc((size_t)L * D * 2);            // 16 MB
    _Float16* WqT  = (_Float16*)alloc((size_t)NQ  * HD * D * 2);     //  8 MB [N,K]
    _Float16* WkT  = (_Float16*)alloc((size_t)NKV * HD * D * 2);     //  4 MB [N,K]
    float*    Qf   = (float*)   alloc((size_t)L * NQ  * HD * 4);     // 32 MB
    float*    Kf   = (float*)   alloc((size_t)L * NKV * HD * 4);     // 16 MB
    _Float16* Qh   = (_Float16*)alloc((size_t)L * NQ  * HD * 2);     // 16 MB
    _Float16* Kh   = (_Float16*)alloc((size_t)L * NKV * HD * 2);     //  8 MB
    float*    rowM = (float*)   alloc((size_t)NQ * L * 4);
    float*    rowZ = (float*)   alloc((size_t)NQ * L * 4);
    float*    imp  = (float*)   alloc((size_t)L * 4);
    int*      sel  = (int*)     alloc((size_t)L * 4);

    // 1. weights -> f16, transposed to [N, K]; embeddings -> f16
    transpose_f16_kernel<<<dim3(D / 64, (NQ  * HD) / 64), 256, 0, stream>>>(
        wq, WqT, D, NQ * HD);
    transpose_f16_kernel<<<dim3(D / 64, (NKV * HD) / 64), 256, 0, stream>>>(
        wk, WkT, D, NKV * HD);
    embed_gather_kernel<<<dim3(L, D / 256), 256, 0, stream>>>(ids, table, Eh);

    // 2. projections (f16 WMMA, async double-buffered LDS staging, f32 out)
    gemm_f16_kernel<<<dim3(L / 128, (NQ  * HD) / 128), 256, 0, stream>>>(
        Eh, WqT, Qf, NQ * HD, D);
    gemm_f16_kernel<<<dim3(L / 128, (NKV * HD) / 128), 256, 0, stream>>>(
        Eh, WkT, Kf, NKV * HD, D);

    // 3. RoPE -> f16
    rope_kernel<<<dim3(L, NQ),  HD, 0, stream>>>(Qf, Qh, NQ);
    rope_kernel<<<dim3(L, NKV), HD, 0, stream>>>(Kf, Kh, NKV);

    // 4. attention: two-pass flash softmax -> importance
    zero_f32_kernel<<<L / 256, 256, 0, stream>>>(imp, L);
    attn_stats_kernel<<<dim3(L / 128, NQ), 256, 0, stream>>>(Qh, Kh, rowM, rowZ);
    attn_accum_kernel<<<dim3(L / 128, NQ), 256, 0, stream>>>(Qh, Kh, rowM, rowZ, imp);

    // 5. top-k select + emit
    rank_kernel<<<L / 256, 256, 0, stream>>>(imp, sel);
    emit_kernel<<<1, 256, 0, stream>>>(sel, ids, (int*)d_out, out_size / 2);
}
